// CrossAttention_33913061769426
// MI455X (gfx1250) — matrix operand order repaired and checked by reading, not compiled, not run on hardware
//
#include <hip/hip_runtime.h>
#include <hip/hip_bf16.h>
#include <math.h>

typedef _Float16 half_t;
typedef __attribute__((ext_vector_type(16))) _Float16 v16h;
typedef __attribute__((ext_vector_type(8)))  _Float16 v8h;
typedef __attribute__((ext_vector_type(8)))  float    v8f;
typedef __attribute__((ext_vector_type(4)))  unsigned int u32x4;
typedef __attribute__((ext_vector_type(8)))  int i32x8;
typedef __attribute__((ext_vector_type(4)))  int i32x4;

#define BB 2
#define QL 3072
#define KL 3072
#define CC 128
#define HH 4
#define DH 32
#define MQ (BB*QL)   // 6144 rows per stream

#if defined(__HIP_DEVICE_COMPILE__) && __has_builtin(__builtin_amdgcn_tensor_load_to_lds) && \
    __has_builtin(__builtin_amdgcn_s_wait_tensorcnt)
#define USE_TDM 1
#else
#define USE_TDM 0
#endif

// ---------------------------------------------------------------- WMMA helpers
__device__ inline v8f wmma16(v16h a, v16h b, v8f c) {
  return __builtin_amdgcn_wmma_f32_16x16x32_f16(
      false, a, false, b, (short)0, c, false, false);
}

// A fragment 16x32 f16 from row-major (lda halves). Lane layout used here (contiguous k, not ISA order):
// row = lane%16 ; halves = p[0..15] contiguous with p offset kb = (lane>=16)*16.
__device__ inline v16h load_afrag(const half_t* A, int lda, int row0, int k0, int lane) {
  int row = row0 + (lane & 15);
  int kb  = ((lane >> 4) & 1) << 4;
  const half_t* p = A + (size_t)row * lda + k0 + kb;
  v8h lo = *(const v8h*)(p);
  v8h hi = *(const v8h*)(p + 8);
  v16h a;
#pragma unroll
  for (int j = 0; j < 8; ++j) { a[j] = lo[j]; a[8 + j] = hi[j]; }
  return a;
}

// B fragment 32(K)x16(N): the 16 K-values per lane are CONTIGUOUS in memory.
__device__ inline v16h load_bfrag_c(const half_t* Bm, int ns, int n0, int lane) {
  int n  = n0 + (lane & 15);
  int kb = ((lane >> 4) & 1) << 4;
  return *(const v16h*)(Bm + (size_t)n * ns + kb);
}

// B fragment from pre-packed weights: frag f = (kc/32)*8 + t, lane-major.
__device__ inline v16h load_bfrag_p(const half_t* Bp, int frag, int lane) {
  return *(const v16h*)(Bp + ((size_t)frag << 9) + (lane << 4));
}

__device__ inline float sigm(float x) { return 1.0f / (1.0f + __expf(-x)); }

// ---------------------------------------------------------------- TDM 2D tile load
#if USE_TDM
__device__ inline void tdm_load_2d(unsigned lds_off, const half_t* gaddr,
                                   unsigned tensor_d0, unsigned tensor_d1,
                                   unsigned tile_d0, unsigned tile_d1,
                                   unsigned stride0) {
  unsigned long long ga = (unsigned long long)gaddr;
  u32x4 g0;
  g0[0] = 1u;                                            // count=1 user descriptor
  g0[1] = lds_off;                                       // lds_addr (bytes)
  g0[2] = (unsigned)(ga & 0xFFFFFFFFu);                  // global_addr[31:0]
  g0[3] = (unsigned)((ga >> 32) & 0x01FFFFFFu) | (2u << 30);  // addr[56:32] | type=2
  i32x8 g1;
  g1[0] = (int)(1u << 16);                               // data_size=1 -> 2 bytes
  g1[1] = (int)((tensor_d0 & 0xFFFFu) << 16);            // tensor_dim0[15:0]
  g1[2] = (int)(((tensor_d0 >> 16) & 0xFFFFu) | ((tensor_d1 & 0xFFFFu) << 16));
  g1[3] = (int)(((tensor_d1 >> 16) & 0xFFFFu) | ((tile_d0 & 0xFFFFu) << 16));
  g1[4] = (int)(tile_d1 & 0xFFFFu);                      // tile_dim1 ; tile_dim2=0
  g1[5] = (int)stride0;                                  // tensor_dim0_stride[31:0]
  g1[6] = 0;
  g1[7] = 0;
  i32x4 gz4 = {0, 0, 0, 0};
  i32x8 gz8 = {0, 0, 0, 0, 0, 0, 0, 0};
  __builtin_amdgcn_tensor_load_to_lds(g0, g1, gz4, gz4, gz8, 0);
}
#endif

// ---------------------------------------------------------------- f32 -> f16
__global__ __launch_bounds__(256) void f32_to_f16_kernel(const float* __restrict__ s,
                                                         half_t* __restrict__ d, int n) {
  int i = blockIdx.x * 256 + threadIdx.x;
  if (i < n) d[i] = (half_t)s[i];
}

// Pack a 128x128 f32 weight into WMMA-B fragment order (f16).
__global__ __launch_bounds__(256) void pack_weight_kernel(const float* __restrict__ W,
                                                          half_t* __restrict__ P) {
  int idx = blockIdx.x * 256 + threadIdx.x;  // 16384 elements
  int f    = idx >> 9;          // fragment: kc4*8 + t
  int lane = (idx >> 4) & 31;
  int j    = idx & 15;
  int kc4 = f >> 3, t = f & 7;
  int n = t * 16 + (lane & 15);
  int k = kc4 * 32 + ((lane >> 4) & 1) * 16 + j;
  P[idx] = (half_t)W[k * CC + n];
}

// ---------------------------------------------------------------- LayerNorm (wave per row)
__global__ __launch_bounds__(128) void aln_norm_kernel(
    const float* __restrict__ x, const float* __restrict__ cond,
    const float* __restrict__ gamma, half_t* __restrict__ xn,
    half_t* __restrict__ cn, int M) {
  int lane = threadIdx.x & 31, wid = threadIdx.x >> 5;
  int row = blockIdx.x * 4 + wid;
  if (row >= M) return;
  const float4 xv = ((const float4*)(x    + (size_t)row * CC))[lane];
  const float4 cv = ((const float4*)(cond + (size_t)row * CC))[lane];

  float s = xv.x + xv.y + xv.z + xv.w;
#pragma unroll
  for (int off = 16; off; off >>= 1) s += __shfl_xor(s, off, 32);
  float m = s * (1.0f / CC);
  float d0 = xv.x - m, d1 = xv.y - m, d2 = xv.z - m, d3 = xv.w - m;
  float vs = d0 * d0 + d1 * d1 + d2 * d2 + d3 * d3;
#pragma unroll
  for (int off = 16; off; off >>= 1) vs += __shfl_xor(vs, off, 32);
  float r = rsqrtf(vs * (1.0f / CC) + 1e-5f);
  half_t* xo = xn + (size_t)row * CC + lane * 4;
  xo[0] = (half_t)(d0 * r); xo[1] = (half_t)(d1 * r);
  xo[2] = (half_t)(d2 * r); xo[3] = (half_t)(d3 * r);

  float sc = cv.x + cv.y + cv.z + cv.w;
#pragma unroll
  for (int off = 16; off; off >>= 1) sc += __shfl_xor(sc, off, 32);
  float mc = sc * (1.0f / CC);
  float e0 = cv.x - mc, e1 = cv.y - mc, e2 = cv.z - mc, e3 = cv.w - mc;
  float vc = e0 * e0 + e1 * e1 + e2 * e2 + e3 * e3;
#pragma unroll
  for (int off = 16; off; off >>= 1) vc += __shfl_xor(vc, off, 32);
  float rc = rsqrtf(vc * (1.0f / CC) + 1e-5f);
  const float* g = gamma + lane * 4;
  half_t* co = cn + (size_t)row * CC + lane * 4;
  co[0] = (half_t)(e0 * rc * g[0]); co[1] = (half_t)(e1 * rc * g[1]);
  co[2] = (half_t)(e2 * rc * g[2]); co[3] = (half_t)(e3 * rc * g[3]);
}

// ---------------------------------------------------------------- dual-B WMMA GEMM (packed B)
// MODE 0: ALN   out1_f16 = sigmoid(C1+bias)*aux_f16 + C2            (HASB2)
// MODE 1: QG    out1_f16(q_s[B,H,Q,Dh]) = (C1+bias)*scale ; out2_f32 = sigmoid(C2)
// MODE 2: KV    out1_f16(k_s[B,H,K,Dh]) = C1 ; out2_f16(v_t[B,H,Dh,K]) = C2
// MODE 3: SIG   out2_f32 = sigmoid(C1 + bias)                        (!HASB2)
// MODE 4: MUL   out1_f32 = C1 * aux_f32                              (!HASB2)
template <int MODE, bool HASB2>
__global__ __launch_bounds__(128) void gemm_dual_kernel(
    const half_t* __restrict__ A, const half_t* __restrict__ B1,
    const half_t* __restrict__ B2, const float* __restrict__ bias,
    const void* __restrict__ aux, void* __restrict__ out1,
    void* __restrict__ out2, int M, float scale) {
  int lane = threadIdx.x & 31, wid = threadIdx.x >> 5;
  int row0 = (blockIdx.x * 4 + wid) * 16;
  if (row0 >= M) return;

  const v8f z = {0.f, 0.f, 0.f, 0.f, 0.f, 0.f, 0.f, 0.f};
  v8f acc1[8], acc2[8];
#pragma unroll
  for (int t = 0; t < 8; ++t) { acc1[t] = z; acc2[t] = z; }

#pragma unroll
  for (int kc = 0; kc < CC; kc += 32) {
    v16h a = load_afrag(A, CC, row0, kc, lane);
    int fb = (kc >> 5) * 8;
#pragma unroll
    for (int t = 0; t < 8; ++t) {
      acc1[t] = wmma16(a, load_bfrag_p(B1, fb + t, lane), acc1[t]);
      if (HASB2) acc2[t] = wmma16(a, load_bfrag_p(B2, fb + t, lane), acc2[t]);
    }
  }

  int hi = (lane >> 4) & 1, cl = lane & 15;
#pragma unroll
  for (int t = 0; t < 8; ++t)
#pragma unroll
    for (int v = 0; v < 8; ++v) {
      int row = row0 + v + hi * 8, col = t * 16 + cl;
      if (MODE == 0) {
        const half_t* xn = (const half_t*)aux;
        half_t* o = (half_t*)out1;
        size_t idx = (size_t)row * CC + col;
        float sg = sigm(acc1[t][v] + bias[col]);
        o[idx] = (half_t)(sg * (float)xn[idx] + acc2[t][v]);
      } else if (MODE == 1) {
        half_t* qo = (half_t*)out1;
        float* go = (float*)out2;
        int b = row / QL, qq = row - b * QL, h = col >> 5, dh = col & 31;
        qo[(((size_t)b * HH + h) * QL + qq) * DH + dh] =
            (half_t)((acc1[t][v] + bias[col]) * scale);
        go[(size_t)row * CC + col] = sigm(acc2[t][v]);
      } else if (MODE == 2) {
        half_t* ko = (half_t*)out1;
        half_t* vo = (half_t*)out2;
        int b = row / KL, kk = row - b * KL, h = col >> 5, dh = col & 31;
        ko[(((size_t)b * HH + h) * KL + kk) * DH + dh] = (half_t)acc1[t][v];
        vo[(((size_t)b * HH + h) * DH + dh) * KL + kk] = (half_t)acc2[t][v];
      } else if (MODE == 3) {
        float* o = (float*)out2;
        o[(size_t)row * CC + col] = sigm(acc1[t][v] + bias[col]);
      } else {
        const float* zg = (const float*)aux;
        float* o = (float*)out1;
        size_t idx = (size_t)row * CC + col;
        o[idx] = acc1[t][v] * zg[idx];
      }
    }
}

// ---------------------------------------------------------------- flash attention
// 4 waves/block, all on the same (b,h); K and transposed-V chunks staged in LDS
// (TDM when available), pair_logits streamed non-temporally.
__global__ __launch_bounds__(128) void attn_kernel(
    const half_t* __restrict__ q_s, const half_t* __restrict__ k_s,
    const half_t* __restrict__ v_t, const float* __restrict__ pair_logits,
    const int* __restrict__ mask_q, const int* __restrict__ mask_k,
    const float* __restrict__ gate, half_t* __restrict__ wa) {
  __shared__ __align__(64) half_t Kch[32 * 32];      // [key][dh]
  __shared__ __align__(64) half_t Vch[32 * 32];      // [dh][key]
  __shared__ __align__(64) half_t Pst[4][16][40];    // per-wave P tile

  int lane = threadIdx.x & 31, wid = threadIdx.x >> 5;
  int tile = blockIdx.x * 4 + wid;                   // 4 consecutive q-tiles: same (b,h)
  const int QT = QL / 16;
  int qt = tile % QT;
  int h  = (tile / QT) % HH;
  int b  = tile / (QT * HH);
  int q0 = qt * 16;
  int hi = (lane >> 4) & 1, cl = lane & 15;

  const half_t* qbase  = q_s + (((size_t)b * HH + h) * QL) * DH;
  const half_t* kbase  = k_s + (((size_t)b * HH + h) * KL) * DH;
  const half_t* vtbase = v_t + (((size_t)b * HH + h) * DH) * KL;
  const float*  plbase = pair_logits + (((size_t)b * HH + h) * QL + q0) * KL;

  v16h qf = load_afrag(qbase, DH, q0, 0, lane);      // q pre-scaled by Dh^-0.5

  float mqf[8];
#pragma unroll
  for (int v = 0; v < 8; ++v)
    mqf[v] = (float)mask_q[b * QL + q0 + v + hi * 8] - 1.0f;

  const v8f z = {0.f, 0.f, 0.f, 0.f, 0.f, 0.f, 0.f, 0.f};
  v8f o0 = z, o1 = z;
  float mrow[8], lrow[8];
#pragma unroll
  for (int v = 0; v < 8; ++v) { mrow[v] = -1e30f; lrow[v] = 0.0f; }

#if USE_TDM
  unsigned koff = (unsigned)(unsigned long long)(__attribute__((address_space(3))) half_t*)Kch;
  unsigned voff = (unsigned)(unsigned long long)(__attribute__((address_space(3))) half_t*)Vch;
#endif

  for (int kb = 0; kb < KL; kb += 32) {
#if USE_TDM
    if (wid == 0) {
      // K chunk: tensor [KL][32dh] stride 32, tile 32x32 starting at row kb
      tdm_load_2d(koff, kbase + (size_t)kb * DH, 32u, (unsigned)KL, 32u, 32u, 32u);
      // Vt chunk: tensor [32dh][KL] stride KL, tile 32keys x 32dh starting at col kb
      tdm_load_2d(voff, vtbase + (size_t)kb, (unsigned)KL, 32u, 32u, 32u, (unsigned)KL);
      __builtin_amdgcn_s_wait_tensorcnt(0);
    }
#else
    {
      int tt = threadIdx.x;
      *(v8h*)(Kch + tt * 8) = *(const v8h*)(kbase + (size_t)kb * DH + tt * 8);
      int dh = tt >> 2, kg = (tt & 3) << 3;
      *(v8h*)(Vch + dh * 32 + kg) = *(const v8h*)(vtbase + (size_t)dh * KL + kb + kg);
    }
#endif
    __syncthreads();

    // S = Q @ K^T : two 16-key tiles from LDS
    v8f s0 = wmma16(qf, load_bfrag_c(Kch, 32, 0, lane), z);
    v8f s1 = wmma16(qf, load_bfrag_c(Kch, 32, 16, lane), z);

    int key0 = kb + cl;
    float mk0 = (float)mask_k[b * KL + key0] - 1.0f;
    float mk1 = (float)mask_k[b * KL + key0 + 16] - 1.0f;

#pragma unroll
    for (int v = 0; v < 8; ++v) {
      int row = v + hi * 8;
      const float* pl = plbase + (size_t)row * KL + kb + cl;
      float l0 = s0[v] + __builtin_nontemporal_load(pl)      + 1e9f * mqf[v] * mk0;
      float l1 = s1[v] + __builtin_nontemporal_load(pl + 16) + 1e9f * mqf[v] * mk1;
      float mx = fmaxf(l0, l1);
#pragma unroll
      for (int off = 1; off < 16; off <<= 1) mx = fmaxf(mx, __shfl_xor(mx, off, 32));
      float mn = fmaxf(mrow[v], mx);
      float corr = __expf(mrow[v] - mn);
      mrow[v] = mn;
      float e0 = __expf(l0 - mn), e1 = __expf(l1 - mn);
      float rs = e0 + e1;
#pragma unroll
      for (int off = 1; off < 16; off <<= 1) rs += __shfl_xor(rs, off, 32);
      lrow[v] = lrow[v] * corr + rs;
      o0[v] *= corr; o1[v] *= corr;
      Pst[wid][row][cl]      = (half_t)e0;
      Pst[wid][row][cl + 16] = (half_t)e1;
    }
    asm volatile("s_wait_dscnt 0" ::: "memory");

    // O += P @ V
    v16h pf = load_afrag(&Pst[wid][0][0], 40, 0, 0, lane);
    o0 = wmma16(pf, load_bfrag_c(Vch, 32, 0, lane), o0);
    o1 = wmma16(pf, load_bfrag_c(Vch, 32, 16, lane), o1);
    __syncthreads();   // before next chunk overwrites Kch/Vch
  }

#pragma unroll
  for (int v = 0; v < 8; ++v) {
    int row = q0 + v + hi * 8;
    float inv = 1.0f / lrow[v];
    size_t gb = ((size_t)b * QL + row) * CC + h * DH;
    float g0 = gate[gb + cl], g1 = gate[gb + 16 + cl];
    wa[gb + cl]      = (half_t)(o0[v] * inv * g0);
    wa[gb + 16 + cl] = (half_t)(o1[v] * inv * g1);
  }
}

// ---------------------------------------------------------------- host
extern "C" void kernel_launch(void* const* d_in, const int* in_sizes, int n_in,
                              void* d_out, int out_size, void* d_ws, size_t ws_size,
                              hipStream_t stream) {
  const float* x_q  = (const float*)d_in[0];
  const float* x_k  = (const float*)d_in[1];
  const int*   mask_q = (const int*)d_in[2];
  const int*   mask_k = (const int*)d_in[3];
  const float* pair_logits = (const float*)d_in[4];
  const float* sc_q = (const float*)d_in[5];
  const float* sc_k = (const float*)d_in[6];
  const float* gamma_cq = (const float*)d_in[7];
  const float* Wsq = (const float*)d_in[8];
  const float* bsq = (const float*)d_in[9];
  const float* Wbq = (const float*)d_in[10];
  const float* gamma_ck = (const float*)d_in[11];
  const float* Wsk = (const float*)d_in[12];
  const float* bsk = (const float*)d_in[13];
  const float* Wbk = (const float*)d_in[14];
  const float* Wq  = (const float*)d_in[15];
  const float* bq  = (const float*)d_in[16];
  const float* Wk  = (const float*)d_in[17];
  const float* Wv  = (const float*)d_in[18];
  const float* Wg  = (const float*)d_in[19];
  const float* Wt2 = (const float*)d_in[20];
  const float* Wzc = (const float*)d_in[21];
  const float* bzc = (const float*)d_in[22];
  (void)in_sizes; (void)n_in; (void)out_size; (void)ws_size;

  char* w = (char*)d_ws;
  auto alloc = [&](size_t bytes) { char* p = w; w += (bytes + 255) & ~(size_t)255; return p; };
  const size_t WSZ = (size_t)CC * CC * sizeof(half_t);
  const size_t ASZ = (size_t)MQ * CC * sizeof(half_t);
  const size_t FSZ = (size_t)MQ * CC * sizeof(float);

  half_t* wWsq = (half_t*)alloc(WSZ); half_t* wWbq = (half_t*)alloc(WSZ);
  half_t* wWsk = (half_t*)alloc(WSZ); half_t* wWbk = (half_t*)alloc(WSZ);
  half_t* wWq  = (half_t*)alloc(WSZ); half_t* wWk  = (half_t*)alloc(WSZ);
  half_t* wWv  = (half_t*)alloc(WSZ); half_t* wWg  = (half_t*)alloc(WSZ);
  half_t* wWt2 = (half_t*)alloc(WSZ); half_t* wWzc = (half_t*)alloc(WSZ);
  half_t* xnq = (half_t*)alloc(ASZ);  half_t* cnq = (half_t*)alloc(ASZ);
  half_t* xnk = (half_t*)alloc(ASZ);  half_t* cnk = (half_t*)alloc(ASZ);
  half_t* scq16 = (half_t*)alloc(ASZ);
  half_t* xq16 = (half_t*)alloc(ASZ); half_t* xk16 = (half_t*)alloc(ASZ);
  half_t* q_s = (half_t*)alloc(ASZ);  half_t* k_s = (half_t*)alloc(ASZ);
  half_t* v_t = (half_t*)alloc(ASZ);
  float*  gate = (float*)alloc(FSZ);  float* zg = (float*)alloc(FSZ);
  half_t* wa16 = (half_t*)alloc(ASZ);

  const int WN = CC * CC;
  const float* wsrc[10] = {Wsq, Wbq, Wsk, Wbk, Wq, Wk, Wv, Wg, Wt2, Wzc};
  half_t* wdst[10] = {wWsq, wWbq, wWsk, wWbk, wWq, wWk, wWv, wWg, wWt2, wWzc};
  for (int i = 0; i < 10; ++i)
    pack_weight_kernel<<<WN / 256, 256, 0, stream>>>(wsrc[i], wdst[i]);
  f32_to_f16_kernel<<<(MQ * CC + 255) / 256, 256, 0, stream>>>(sc_q, scq16, MQ * CC);

  aln_norm_kernel<<<MQ / 4, 128, 0, stream>>>(x_q, sc_q, gamma_cq, xnq, cnq, MQ);
  aln_norm_kernel<<<MQ / 4, 128, 0, stream>>>(x_k, sc_k, gamma_ck, xnk, cnk, MQ);

  const int GG = MQ / 64;
  gemm_dual_kernel<0, true><<<GG, 128, 0, stream>>>(cnq, wWsq, wWbq, bsq, xnq, xq16, nullptr, MQ, 0.f);
  gemm_dual_kernel<0, true><<<GG, 128, 0, stream>>>(cnk, wWsk, wWbk, bsk, xnk, xk16, nullptr, MQ, 0.f);
  gemm_dual_kernel<1, true><<<GG, 128, 0, stream>>>(xq16, wWq, wWg, bq, nullptr, q_s, gate, MQ,
                                                    0.17677669529663687f);
  gemm_dual_kernel<2, true><<<GG, 128, 0, stream>>>(xk16, wWk, wWv, nullptr, nullptr, k_s, v_t, MQ, 0.f);
  gemm_dual_kernel<3, false><<<GG, 128, 0, stream>>>(scq16, wWzc, nullptr, bzc, nullptr, nullptr, zg, MQ, 0.f);

  attn_kernel<<<(BB * HH * (QL / 16)) / 4, 128, 0, stream>>>(
      q_s, k_s, v_t, pair_logits, mask_q, mask_k, gate, wa16);

  gemm_dual_kernel<4, false><<<GG, 128, 0, stream>>>(wa16, wWt2, nullptr, nullptr, zg, d_out, nullptr, MQ, 0.f);
}
